// NIC_83030307766754
// MI455X (gfx1250) — matrix operand order repaired and checked by reading, not compiled, not run on hardware
//
#include <hip/hip_runtime.h>
#include <hip/hip_bf16.h>
#include <cstdint>

// Problem dimensions (fixed by the reference)
#define Bc   128
#define Tc   20
#define Gc   41
#define DINc 1536
#define Ec   512
#define ETc  512
#define Uc   512
#define Ac   256
#define Vc   8000

typedef __attribute__((ext_vector_type(16))) __bf16 bf16x16;
typedef __attribute__((ext_vector_type(8)))  float  f32x8;

// GEMM tiling: workgroup = 256 threads = 8 waves; tile 64x128; K-chunk 32.
// Each wave computes a 32x32 output tile via 4x v_wmma_f32_16x16x32_bf16.
#define TM  64
#define TN  128
#define TK  32
#define TKP 34    // LDS row stride for A tile (bf16 elems) -> 68B, odd-bank
#define TNP 144   // LDS row stride for B tile -> 288B = 72 DW (16B aligned rows)

__device__ __forceinline__ __bf16 to_bf16(float f) {
  unsigned u = __builtin_bit_cast(unsigned, f);
  u += 0x7fffu + ((u >> 16) & 1u);            // round-to-nearest-even
  unsigned short h = (unsigned short)(u >> 16);
  return __builtin_bit_cast(__bf16, h);
}

__device__ __forceinline__ __bf16 bf16_zero() {
  return __builtin_bit_cast(__bf16, (unsigned short)0);
}

__device__ __forceinline__ float leaky02(float v) { return v >= 0.f ? v : 0.2f * v; }

// LDS byte offset of a generic (flat) shared pointer: ISA 10.2 -> addr[31:0]
__device__ __forceinline__ unsigned lds_off_u32(const void* p) {
  return (unsigned)(uintptr_t)p;
}

// C[batch] = act( A[batch](MxK,fp32) * B[batch](KxN,bf16) + bias[batch] )
// A converted fp32->bf16 during LDS staging; B staged via async global->LDS DMA.
__global__ __launch_bounds__(256)
void gemm_bf16_wmma_kernel(const float* __restrict__ A, long long aBS, int lda,
                           const __bf16* __restrict__ Bm, long long bBS, int ldb,
                           const float* __restrict__ bias, long long biasBS,
                           float* __restrict__ C, long long cBS, int ldc,
                           int M, int N, int K, int act)
{
  __shared__ __align__(16) __bf16 As[TM * TKP];
  __shared__ __align__(16) __bf16 Bs[TK * TNP];

  const int tid  = threadIdx.x;
  const int lane = tid & 31;
  const int wid  = tid >> 5;
  const int wm   = wid >> 2;     // 0..1 -> row sub-tile of 32
  const int wn   = wid & 3;      // 0..3 -> col sub-tile of 32
  const int hlf  = lane >> 4;    // lane half (ISA fragment layout)
  const int mrow = lane & 15;
  const int ncol = lane & 15;

  const int mBlk = blockIdx.y * TM;
  const int nBlk = blockIdx.x * TN;

  const float*  Ab    = A  + (long long)blockIdx.z * aBS;
  const __bf16* Bb    = Bm + (long long)blockIdx.z * bBS;
  const float*  biasb = bias ? bias + (long long)blockIdx.z * biasBS : nullptr;
  float*        Cb    = C  + (long long)blockIdx.z * cBS;

  f32x8 acc00 = {}; f32x8 acc01 = {}; f32x8 acc10 = {}; f32x8 acc11 = {};

  for (int k0 = 0; k0 < K; k0 += TK) {
    // ---- Stage B tile (TK x TN, bf16) via async global->LDS B128 copies ----
    // 512 chunks of 16B; 256 threads -> 2 chunks each.
    for (int i = tid; i < (TK * TN) / 8; i += 256) {
      int k  = i >> 4;          // TN/8 = 16 chunks per row
      int n8 = i & 15;
      int gn = nBlk + n8 * 8;
      const __bf16* gp = Bb + (size_t)(k0 + k) * ldb + gn;
      __bf16* lp = &Bs[k * TNP + n8 * 8];
      if (gn + 8 <= N) {
        unsigned lds = lds_off_u32(lp);
        unsigned long long ga = (unsigned long long)(uintptr_t)gp;
        asm volatile("global_load_async_to_lds_b128 %0, %1, off"
                     :: "v"(lds), "v"(ga) : "memory");
      } else {
        #pragma unroll
        for (int e = 0; e < 8; ++e)
          lp[e] = (gn + e < N) ? gp[e] : bf16_zero();
      }
    }
    // ---- Stage A tile (TM x TK), fp32 -> bf16 ----
    for (int i = tid; i < TM * TK; i += 256) {
      int m = i / TK, k = i % TK;
      int gm = mBlk + m;
      float v = (gm < M) ? Ab[(long long)gm * lda + (k0 + k)] : 0.f;
      As[m * TKP + k] = to_bf16(v);
    }
    // Prefetch next K-chunk of A (speculative; dropped if invalid)
    if (k0 + TK < K) {
      __builtin_prefetch(&Ab[(long long)(mBlk + (tid & 63)) * lda + (k0 + TK)], 0, 1);
      __builtin_prefetch(&Bb[(long long)(k0 + TK + (tid >> 3)) * ldb + nBlk + (tid & 7) * 16], 0, 1);
    }
    // All of this wave's async DMAs must have landed in LDS before the barrier.
    asm volatile("s_wait_asynccnt 0x0" ::: "memory");
    __syncthreads();

    // ---- Build fragments per ISA 7.12.2 layouts (wave32) ----
    bf16x16 a0, a1, b0, b1;
    {
      const __bf16* r0 = &As[(wm * 32 +  0 + mrow) * TKP];
      const __bf16* r1 = &As[(wm * 32 + 16 + mrow) * TKP];
      #pragma unroll
      for (int v = 0; v < 8; ++v) {
        int kb = (v < 4) ? (hlf * 8 + v * 2) : (16 + hlf * 8 + (v - 4) * 2);
        a0[2 * v]     = r0[kb];
        a0[2 * v + 1] = r0[kb + 1];
        a1[2 * v]     = r1[kb];
        a1[2 * v + 1] = r1[kb + 1];
      }
      const __bf16* br = &Bs[wn * 32 + ncol];
      #pragma unroll
      for (int e = 0; e < 16; ++e) { int kk = (e < 8) ? (hlf * 8 + e) : (16 + hlf * 8 + (e - 8)); b0[e] = br[kk * TNP]; b1[e] = br[kk * TNP + 16]; }
    }

    acc00 = __builtin_amdgcn_wmma_f32_16x16x32_bf16(false, a0, false, b0, (short)0, acc00, false, false);
    acc01 = __builtin_amdgcn_wmma_f32_16x16x32_bf16(false, a0, false, b1, (short)0, acc01, false, false);
    acc10 = __builtin_amdgcn_wmma_f32_16x16x32_bf16(false, a1, false, b0, (short)0, acc10, false, false);
    acc11 = __builtin_amdgcn_wmma_f32_16x16x32_bf16(false, a1, false, b1, (short)0, acc11, false, false);
    __syncthreads();
  }

  // Epilogue: C/D layout -> lane (n = lane&15), VGPR r -> m = r + 8*(lane>>4)
  auto store_tile = [&](const f32x8& acc, int ms, int ns) {
    int n = nBlk + wn * 32 + ns + ncol;
    if (n >= N) return;
    float bv = biasb ? biasb[n] : 0.f;
    #pragma unroll
    for (int r = 0; r < 8; ++r) {
      int m = mBlk + wm * 32 + ms + r + 8 * hlf;
      if (m < M) {
        float v = acc[r] + bv;
        if (act) v = leaky02(v);
        Cb[(long long)m * ldc + n] = v;
      }
    }
  };
  store_tile(acc00,  0,  0);
  store_tile(acc01,  0, 16);
  store_tile(acc10, 16,  0);
  store_tile(acc11, 16, 16);
}

// One-time fp32 -> bf16 weight conversion
__global__ __launch_bounds__(256)
void f32_to_bf16_kernel(const float* __restrict__ src, __bf16* __restrict__ dst, long long n)
{
  long long i = (long long)blockIdx.x * 256 + threadIdx.x;
  if (i < n) dst[i] = to_bf16(src[i]);
}

// Per-step attention: scores over G, softmax, context, x = [context, emb]
__global__ __launch_bounds__(256)
void attn_context_kernel(const float* __restrict__ feat_proj, const float* __restrict__ q,
                         const float* __restrict__ b_attn, const float* __restrict__ V_attn,
                         const float* __restrict__ features, const float* __restrict__ emb,
                         const int* __restrict__ text, int t,
                         float* __restrict__ x, float* __restrict__ attn_out)
{
  const int b    = blockIdx.x;
  const int tid  = threadIdx.x;
  const int lane = tid & 31;
  const int wid  = tid >> 5;
  __shared__ float e_s[Gc];
  __shared__ float m_s, sum_s;

  for (int g = wid; g < Gc; g += 8) {
    float p = 0.f;
    const float* fp = feat_proj + ((size_t)b * Gc + g) * Ac;
    const float* qb = q + (size_t)b * Ac;
    for (int a = lane; a < Ac; a += 32) {
      float v = leaky02(fp[a] + qb[a] + b_attn[a]);
      p += v * V_attn[a];
    }
    #pragma unroll
    for (int off = 16; off > 0; off >>= 1) p += __shfl_xor(p, off, 32);
    if (lane == 0) e_s[g] = p;
  }
  __syncthreads();

  if (tid == 0) {
    float mx = -3.4e38f;
    for (int g = 0; g < Gc; ++g) mx = fmaxf(mx, e_s[g]);
    float s = 0.f;
    for (int g = 0; g < Gc; ++g) s += __expf(e_s[g] - mx);
    m_s = mx; sum_s = s;
  }
  __syncthreads();

  if (tid < Gc) {
    float al = __expf(e_s[tid] - m_s) / sum_s;
    e_s[tid] = al;
    attn_out[((size_t)b * Tc + t) * Gc + tid] = al;
  }
  __syncthreads();

  float* xb = x + (size_t)b * (Ec + ETc);
  for (int e = tid; e < Ec; e += 256) {
    float c = 0.f;
    for (int g = 0; g < Gc; ++g)
      c += e_s[g] * features[((size_t)b * Gc + g) * Ec + e];
    xb[e] = c;
  }
  int w = text[b * Tc + t];
  const float* we = emb + (size_t)w * ETc;
  for (int e = tid; e < ETc; e += 256) xb[Ec + e] = we[e];
}

// GRU gates, reset_after=True (biases already folded into xm/rm by GEMM)
__global__ __launch_bounds__(256)
void gru_gate_kernel(const float* __restrict__ xm, const float* __restrict__ rm,
                     const float* __restrict__ h, float* __restrict__ hn)
{
  int i = blockIdx.x * 256 + threadIdx.x;
  if (i >= Bc * Uc) return;
  int b = i / Uc, u = i % Uc;
  const float* xr = xm + (size_t)b * 3 * Uc;
  const float* rr = rm + (size_t)b * 3 * Uc;
  float z  = 1.f / (1.f + __expf(-(xr[u] + rr[u])));
  float r  = 1.f / (1.f + __expf(-(xr[Uc + u] + rr[Uc + u])));
  float hh = tanhf(xr[2 * Uc + u] + r * rr[2 * Uc + u]);
  hn[i] = z * h[i] + (1.f - z) * hh;
}

// Softmax over vocab, writes into outputs[b, t, :]
__global__ __launch_bounds__(256)
void softmax_vocab_kernel(const float* __restrict__ logits, float* __restrict__ out, int t)
{
  const int b = blockIdx.x, tid = threadIdx.x;
  __shared__ float red[256];
  const float* row = logits + (size_t)b * Vc;

  float m = -3.4e38f;
  for (int v = tid; v < Vc; v += 256) m = fmaxf(m, row[v]);
  red[tid] = m; __syncthreads();
  for (int s = 128; s > 0; s >>= 1) { if (tid < s) red[tid] = fmaxf(red[tid], red[tid + s]); __syncthreads(); }
  m = red[0]; __syncthreads();

  float sum = 0.f;
  for (int v = tid; v < Vc; v += 256) sum += __expf(row[v] - m);
  red[tid] = sum; __syncthreads();
  for (int s = 128; s > 0; s >>= 1) { if (tid < s) red[tid] += red[tid + s]; __syncthreads(); }
  float inv = 1.f / red[0];

  float* o = out + ((size_t)b * Tc + t) * Vc;
  for (int v = tid; v < Vc; v += 256) o[v] = __expf(row[v] - m) * inv;
}

__global__ __launch_bounds__(256)
void copy_f32_kernel(const float* __restrict__ src, float* __restrict__ dst, int n)
{
  int i = blockIdx.x * 256 + threadIdx.x;
  if (i < n) dst[i] = src[i];
}

static void launch_gemm(hipStream_t stream,
                        const float* A, long long aBS, int lda,
                        const __bf16* Bm, long long bBS, int ldb,
                        const float* bias, long long biasBS,
                        float* C, long long cBS, int ldc,
                        int M, int N, int K, int act, int batch)
{
  dim3 grid((N + TN - 1) / TN, (M + TM - 1) / TM, batch);
  gemm_bf16_wmma_kernel<<<grid, 256, 0, stream>>>(A, aBS, lda, Bm, bBS, ldb,
                                                  bias, biasBS, C, cBS, ldc,
                                                  M, N, K, act);
}

static void launch_cvt(hipStream_t stream, const float* src, __bf16* dst, long long n)
{
  f32_to_bf16_kernel<<<(unsigned)((n + 255) / 256), 256, 0, stream>>>(src, dst, n);
}

extern "C" void kernel_launch(void* const* d_in, const int* in_sizes, int n_in,
                              void* d_out, int out_size, void* d_ws, size_t ws_size,
                              hipStream_t stream)
{
  (void)in_sizes; (void)n_in; (void)out_size; (void)ws_size;

  const float* img    = (const float*)d_in[0];
  const int*   text   = (const int*)  d_in[1];
  const float* hidden = (const float*)d_in[2];
  /* carry d_in[3] unused by the reference */
  const float* emb    = (const float*)d_in[4];
  const float* W_in   = (const float*)d_in[5];
  const float* b_in   = (const float*)d_in[6];
  const float* W1     = (const float*)d_in[7];
  const float* W2     = (const float*)d_in[8];
  const float* b_attn = (const float*)d_in[9];
  const float* V_attn = (const float*)d_in[10];
  const float* gk     = (const float*)d_in[11];
  const float* gr     = (const float*)d_in[12];
  const float* gb     = (const float*)d_in[13];
  const float* Wi     = (const float*)d_in[14];
  const float* bi     = (const float*)d_in[15];
  const float* Wo     = (const float*)d_in[16];
  const float* bo     = (const float*)d_in[17];

  float* outputs = (float*)d_out;                         // (B,T,V)
  float* attn    = outputs + (size_t)Bc * Tc * Vc;        // (B,T,G,1)

  // ---- workspace carve-out (256B-aligned slabs) ----
  char* wsb = (char*)d_ws;
  auto alloc = [&](size_t bytes) -> void* {
    void* p = (void*)wsb;
    wsb += (bytes + 255) & ~(size_t)255;
    return p;
  };
  // bf16 weight copies (converted once, reused 20x by the scan loop)
  __bf16* W_in_h = (__bf16*)alloc((size_t)Gc * DINc * Ec * 2);
  __bf16* W1_h   = (__bf16*)alloc((size_t)Ec * Ac * 2);
  __bf16* W2_h   = (__bf16*)alloc((size_t)Uc * Ac * 2);
  __bf16* gk_h   = (__bf16*)alloc((size_t)(Ec + ETc) * 3 * Uc * 2);
  __bf16* gr_h   = (__bf16*)alloc((size_t)Uc * 3 * Uc * 2);
  __bf16* Wi_h   = (__bf16*)alloc((size_t)Uc * 256 * 2);
  __bf16* Wo_h   = (__bf16*)alloc((size_t)256 * Vc * 2);
  // fp32 activations
  float* features  = (float*)alloc((size_t)Bc * Gc * Ec * 4);
  float* feat_proj = (float*)alloc((size_t)Bc * Gc * Ac * 4);
  float* h0        = (float*)alloc((size_t)Bc * Uc * 4);
  float* h1        = (float*)alloc((size_t)Bc * Uc * 4);
  float* q         = (float*)alloc((size_t)Bc * Ac * 4);
  float* x         = (float*)alloc((size_t)Bc * (Ec + ETc) * 4);
  float* xm        = (float*)alloc((size_t)Bc * 3 * Uc * 4);
  float* rm        = (float*)alloc((size_t)Bc * 3 * Uc * 4);
  float* sbuf      = (float*)alloc((size_t)Bc * 256 * 4);
  float* logits    = (float*)alloc((size_t)Bc * Vc * 4);

  // ---- one-time weight conversion fp32 -> bf16 ----
  launch_cvt(stream, W_in, W_in_h, (long long)Gc * DINc * Ec);
  launch_cvt(stream, W1,   W1_h,   (long long)Ec * Ac);
  launch_cvt(stream, W2,   W2_h,   (long long)Uc * Ac);
  launch_cvt(stream, gk,   gk_h,   (long long)(Ec + ETc) * 3 * Uc);
  launch_cvt(stream, gr,   gr_h,   (long long)Uc * 3 * Uc);
  launch_cvt(stream, Wi,   Wi_h,   (long long)Uc * 256);
  launch_cvt(stream, Wo,   Wo_h,   (long long)256 * Vc);

  // features[b,g,:] = leaky(img[b,g,:] @ W_in[g] + b_in[g]), batched over G
  launch_gemm(stream,
              img,    (long long)DINc, Gc * DINc,
              W_in_h, (long long)DINc * Ec, Ec,
              b_in,   (long long)Ec,
              features, (long long)Ec, Gc * Ec,
              Bc, Ec, DINc, /*act=*/1, /*batch=*/Gc);

  // feat_proj = features(B*G,E) @ W1
  launch_gemm(stream, features, 0, Ec, W1_h, 0, Ac, nullptr, 0,
              feat_proj, 0, Ac, Bc * Gc, Ac, Ec, 0, 1);

  // h0 = hidden (zeros input; workspace is poisoned so copy explicitly)
  copy_f32_kernel<<<(Bc * Uc + 255) / 256, 256, 0, stream>>>(hidden, h0, Bc * Uc);

  for (int t = 0; t < Tc; ++t) {
    float* hc = (t & 1) ? h1 : h0;
    float* hn = (t & 1) ? h0 : h1;

    // q = h @ W2
    launch_gemm(stream, hc, 0, Uc, W2_h, 0, Ac, nullptr, 0, q, 0, Ac,
                Bc, Ac, Uc, 0, 1);

    // attention scores/softmax/context + x = [context, emb[text[:,t]]]
    attn_context_kernel<<<Bc, 256, 0, stream>>>(feat_proj, q, b_attn, V_attn,
                                                features, emb, text, t, x, attn);

    // xm = x @ gru_kernel + gru_bias[0]
    launch_gemm(stream, x, 0, Ec + ETc, gk_h, 0, 3 * Uc, gb, 0,
                xm, 0, 3 * Uc, Bc, 3 * Uc, Ec + ETc, 0, 1);

    // rm = h @ gru_rec + gru_bias[1]
    launch_gemm(stream, hc, 0, Uc, gr_h, 0, 3 * Uc, gb + 3 * Uc, 0,
                rm, 0, 3 * Uc, Bc, 3 * Uc, Uc, 0, 1);

    // GRU gates -> h_new
    gru_gate_kernel<<<(Bc * Uc + 255) / 256, 256, 0, stream>>>(xm, rm, hc, hn);

    // s = leaky(h_new @ Wi + bi)
    launch_gemm(stream, hn, 0, Uc, Wi_h, 0, 256, bi, 0, sbuf, 0, 256,
                Bc, 256, Uc, 1, 1);

    // logits = s @ Wo + bo
    launch_gemm(stream, sbuf, 0, 256, Wo_h, 0, Vc, bo, 0, logits, 0, Vc,
                Bc, Vc, 256, 0, 1);

    // outputs[:, t, :] = softmax(logits)
    softmax_vocab_kernel<<<Bc, 256, 0, stream>>>(logits, outputs, t);
  }
}
